// AttentivePosteriorC_15814069584295
// MI455X (gfx1250) — matrix operand order repaired and checked by reading, not compiled, not run on hardware
//
#include <hip/hip_runtime.h>
#include <hip/hip_bf16.h>
#include <math.h>

#define HIDC   512
#define NHEADC 4
#define HDC    128
#define BSC    32
#define NSC    128
#define PC     16      // HW*HW

typedef __attribute__((ext_vector_type(16))) __bf16 v16bf;
typedef __attribute__((ext_vector_type(8)))  float  v8f;

__device__ __forceinline__ unsigned short f2bf(float f) {
  unsigned int u = __builtin_bit_cast(unsigned int, f);
  u += 0x7fffu + ((u >> 16) & 1u);            // round-to-nearest-even
  return (unsigned short)(u >> 16);
}
__device__ __forceinline__ float bf2f(unsigned short h) {
  unsigned int u = ((unsigned int)h) << 16;
  return __builtin_bit_cast(float, u);
}

struct B32x2 { uint4 lo, hi; };
__device__ __forceinline__ v16bf frag16(const unsigned short* p0, const unsigned short* p1) {
  B32x2 u; u.lo = *(const uint4*)p0; u.hi = *(const uint4*)p1;
  return __builtin_bit_cast(v16bf, u);
}

#define TM 128
#define TN 128
#define TK 32
#define LDT 40   // padded LDS row (ushorts)

// Accumulate acc += W[m0:m0+128, :K] @ X[:, c0:c0+128]  (bf16 WMMA, f32 accum)
// X element (k, col=(n,p)) at X + n*xStrideN + k*PC + p
__device__ __forceinline__ void gemm_accum(
    const float* __restrict__ W, const float* __restrict__ X, int K, int ldW,
    long xStrideN, unsigned short* Ash, unsigned short* Bsh,
    v8f acc[8], int m0, int c0)
{
  const int tid = threadIdx.x, lane = tid & 31, wave = tid >> 5;
  for (int k0 = 0; k0 < K; k0 += TK) {
    { // stage A: 128 rows x 32 K of W, f32 -> bf16
      int row = tid & 127, kseg = (tid >> 7) * 16;
      const float* wr = W + (long)(m0 + row) * ldW + k0 + kseg;
      unsigned short* dst = Ash + row * LDT + kseg;
#pragma unroll
      for (int i = 0; i < 4; ++i) {
        float4 f = ((const float4*)wr)[i];
        uint2 pk; pk.x = f2bf(f.x) | ((unsigned)f2bf(f.y) << 16);
                  pk.y = f2bf(f.z) | ((unsigned)f2bf(f.w) << 16);
        *(uint2*)(dst + i * 4) = pk;
      }
    }
    { // stage B: 32 K x 128 cols of X, transposed to [col][k] bf16
      int colL = tid & 127, kseg = (tid >> 7) * 16;
      int col = c0 + colL, nidx = col >> 4, p = col & 15;
      const float* xp = X + (long)nidx * xStrideN + (long)(k0 + kseg) * PC + p;
      unsigned short* dst = Bsh + colL * LDT + kseg;
#pragma unroll
      for (int i = 0; i < 4; ++i) {
        float e0 = xp[(i*4+0)*PC], e1 = xp[(i*4+1)*PC];
        float e2 = xp[(i*4+2)*PC], e3 = xp[(i*4+3)*PC];
        uint2 pk; pk.x = f2bf(e0) | ((unsigned)f2bf(e1) << 16);
                  pk.y = f2bf(e2) | ((unsigned)f2bf(e3) << 16);
        *(uint2*)(dst + i * 4) = pk;
      }
    }
    __syncthreads();
    { // 8 WMMAs per A-fragment (wave owns 16-row strip, 8 N-subtiles)
      const unsigned short* arow = Ash + (wave * 16 + (lane & 15)) * LDT;
      const int kh = (lane >> 4) * 8;           // A: lanes 0-15 K0-7/16-23, 16-31 K8-15/24-31
      v16bf af = frag16(arow + kh, arow + 16 + kh);
      const int kb = (lane >> 4) * 8;           // B: lanes 0-15 K0-7/16-23, 16-31 K8-15/24-31
#pragma unroll
      for (int s = 0; s < 8; ++s) {
        const unsigned short* bcol = Bsh + (s * 16 + (lane & 15)) * LDT + kb;
        v16bf bf = frag16(bcol, bcol + 16);
        acc[s] = __builtin_amdgcn_wmma_f32_16x16x32_bf16(
            false, af, false, bf, (short)0, acc[s], false, false);
      }
    }
    __syncthreads();
  }
}

// Generic GEMM: Y = act( W0@X0 [+ W1@X1] + bias[d] [+ addvec[b][d][p]] )
template<int WRITE_BF16, int ACT>
__global__ __launch_bounds__(256)
void wmma_gemm(const float* __restrict__ W0, const float* __restrict__ X0, int K0,
               const float* __restrict__ W1, const float* __restrict__ X1, int K1,
               int ldW, long xStrideN,
               const float* __restrict__ bias,
               const float* __restrict__ addvec, int nsGroup, int addvecLD,
               void* __restrict__ Yv, long yStrideN)
{
  __shared__ unsigned short Ash[TM * LDT];
  __shared__ unsigned short Bsh[TN * LDT];
  const int lane = threadIdx.x & 31, wave = threadIdx.x >> 5;
  const int m0 = blockIdx.y * TM, c0 = blockIdx.x * TN;

  v8f acc[8];
  const v8f zero = {0.f,0.f,0.f,0.f,0.f,0.f,0.f,0.f};
#pragma unroll
  for (int s = 0; s < 8; ++s) acc[s] = zero;

  gemm_accum(W0, X0, K0, ldW, xStrideN, Ash, Bsh, acc, m0, c0);
  if (K1 > 0) gemm_accum(W1, X1, K1, ldW, xStrideN, Ash, Bsh, acc, m0, c0);

  const int mrow = (lane >> 4) * 8;  // D: lanes 0-15 M=i, lanes 16-31 M=8+i
#pragma unroll
  for (int s = 0; s < 8; ++s) {
    int col = c0 + s * 16 + (lane & 15);
    int nidx = col >> 4, p = col & 15;
#pragma unroll
    for (int i = 0; i < 8; ++i) {
      int d = m0 + wave * 16 + mrow + i;
      float v = acc[s][i];
      if (bias) v += bias[d];
      if (addvec) v += addvec[(long)(nidx / nsGroup) * addvecLD + (long)d * PC + p];
      if (ACT == 1) v = v > 0.f ? v : (expf(v) - 1.f);   // ELU
      long off = (long)nidx * yStrideN + (long)d * PC + p;
      if (WRITE_BF16) ((unsigned short*)Yv)[off] = f2bf(v);
      else            ((float*)Yv)[off] = v;
    }
  }
}

__global__ void vec_add2(const float* a, const float* b, float* o, int n) {
  int i = blockIdx.x * 256 + threadIdx.x;
  if (i < n) o[i] = a[i] + b[i];
}

// zq[b][d][p] = mean_t e[b,t,d,p]
__global__ void zq_reduce_kernel(const unsigned short* __restrict__ e, float* __restrict__ zq) {
  int idx = blockIdx.x * 256 + threadIdx.x;       // BSC*8192
  int b = idx >> 13, dp = idx & 8191;
  const unsigned short* p = e + (long)b * NSC * 8192 + dp;
  float s = 0.f;
  for (int t = 0; t < NSC; ++t) s += bf2f(p[(long)t * 8192]);
  zq[idx] = s * (1.0f / NSC);
}

// qproj[b][n][j][p] = sum_c Wk[n*HD+c][j] * q[b][n*HD+c][p]   (Wk_head^T @ q_head)
__global__ void qproj_kernel(const float* __restrict__ Wk, const float* __restrict__ q,
                             float* __restrict__ qproj) {
  int idx = blockIdx.x * 256 + threadIdx.x;       // 32*4*512*16
  int p = idx & 15, j = (idx >> 4) & 511, n = (idx >> 13) & 3, b = idx >> 15;
  const float* qp = q + (long)b * 8192 + (n * HDC) * PC + p;
  const float* wk = Wk + (long)(n * HDC) * HIDC + j;
  float s = 0.f;
  for (int c = 0; c < HDC; ++c) s += wk[(long)c * HIDC] * qp[c * PC];
  qproj[idx] = s;
}

// qbias[b*4+n] = sum_{c,p} q[b][n*HD+c][p] * bk[n*HD+c]
__global__ void qbias_kernel(const float* __restrict__ q, const float* __restrict__ bk,
                             float* __restrict__ qbias) {
  __shared__ float red[256];
  int bn = blockIdx.x, b = bn >> 2, n = bn & 3;
  float s = 0.f;
  for (int i = threadIdx.x; i < HDC * PC; i += 256) {
    int c = i >> 4, p = i & 15;
    s += q[(long)b * 8192 + (n * HDC + c) * PC + p] * bk[n * HDC + c];
  }
  red[threadIdx.x] = s; __syncthreads();
  for (int st = 128; st > 0; st >>= 1) {
    if (threadIdx.x < st) red[threadIdx.x] += red[threadIdx.x + st];
    __syncthreads();
  }
  if (threadIdx.x == 0) qbias[bn] = red[0];
}

// raw sim[b,n,t] = scale*(<qproj[b,n], e[b,t]> + qbias[b,n])
__global__ void sim_kernel(const unsigned short* __restrict__ e, const float* __restrict__ qproj,
                           const float* __restrict__ qbias, float* __restrict__ att) {
  __shared__ float red[4][256];
  int bt = blockIdx.x, b = bt >> 7, t = bt & 127;
  const unsigned short* ep = e + (long)bt * 8192;
  const float* qp = qproj + (long)b * 32768;
  float a0 = 0, a1 = 0, a2 = 0, a3 = 0;
  for (int i = threadIdx.x; i < 8192; i += 256) {
    float ev = bf2f(ep[i]);
    a0 += ev * qp[i];       a1 += ev * qp[8192 + i];
    a2 += ev * qp[16384 + i]; a3 += ev * qp[24576 + i];
  }
  red[0][threadIdx.x] = a0; red[1][threadIdx.x] = a1;
  red[2][threadIdx.x] = a2; red[3][threadIdx.x] = a3;
  __syncthreads();
  for (int st = 128; st > 0; st >>= 1) {
    if (threadIdx.x < st)
      for (int n = 0; n < 4; ++n) red[n][threadIdx.x] += red[n][threadIdx.x + st];
    __syncthreads();
  }
  if (threadIdx.x < 4) {
    const float scale = 0.022097086912079608f;  // 1/sqrt(HW*HW*hd) = 1/sqrt(2048)
    att[((long)b * 4 + threadIdx.x) * 128 + t] =
        scale * (red[threadIdx.x][0] + qbias[b * 4 + threadIdx.x]);
  }
}

__global__ void softmax_kernel(float* att) {
  __shared__ float red[128];
  int bn = blockIdx.x, t = threadIdx.x;
  float v = att[bn * 128 + t];
  red[t] = v; __syncthreads();
  for (int st = 64; st > 0; st >>= 1) { if (t < st) red[t] = fmaxf(red[t], red[t + st]); __syncthreads(); }
  float m = red[0]; __syncthreads();
  float ex = __expf(v - m);
  red[t] = ex; __syncthreads();
  for (int st = 64; st > 0; st >>= 1) { if (t < st) red[t] += red[t + st]; __syncthreads(); }
  att[bn * 128 + t] = ex / red[0];
}

// ebar[b][n][jp] = sum_t att[b,n,t] * e[b,t,jp]
__global__ void ebar_kernel(const unsigned short* __restrict__ e, const float* __restrict__ att,
                            float* __restrict__ ebar) {
  __shared__ float sw[512];
  int blk = blockIdx.x, b = blk >> 5, chunk = blk & 31;
  sw[threadIdx.x]       = att[(long)b * 512 + threadIdx.x];
  sw[threadIdx.x + 256] = att[(long)b * 512 + threadIdx.x + 256];
  __syncthreads();
  int jp = chunk * 256 + threadIdx.x;
  const unsigned short* ep = e + (long)b * NSC * 8192 + jp;
  float a0 = 0, a1 = 0, a2 = 0, a3 = 0;
  for (int t = 0; t < NSC; ++t) {
    float ev = bf2f(ep[(long)t * 8192]);
    a0 += sw[t] * ev; a1 += sw[128 + t] * ev; a2 += sw[256 + t] * ev; a3 += sw[384 + t] * ev;
  }
  long base = (long)b * 32768 + jp;
  ebar[base] = a0; ebar[base + 8192] = a1; ebar[base + 16384] = a2; ebar[base + 24576] = a3;
}

extern "C" void kernel_launch(void* const* d_in, const int* in_sizes, int n_in,
                              void* d_out, int out_size, void* d_ws, size_t ws_size,
                              hipStream_t stream) {
  (void)in_sizes; (void)n_in; (void)out_size; (void)ws_size;
  const float* h  = (const float*)d_in[0];
  const float* z  = (const float*)d_in[1];
  const float* cc = (const float*)d_in[2];
  const float* Wh = (const float*)d_in[5];
  const float* bh = (const float*)d_in[6];
  const float* Wz = (const float*)d_in[7];
  const float* bz = (const float*)d_in[8];
  const float* Wc = (const float*)d_in[9];
  const float* bc = (const float*)d_in[10];
  const float* Wq = (const float*)d_in[11];
  const float* bq = (const float*)d_in[12];
  const float* Wk = (const float*)d_in[13];
  const float* bk = (const float*)d_in[14];
  const float* Wv = (const float*)d_in[15];
  const float* bv = (const float*)d_in[16];
  const float* Wp = (const float*)d_in[17];
  const float* bp = (const float*)d_in[18];
  const float* Wo = (const float*)d_in[19];
  const float* bo = (const float*)d_in[20];

  float* ws = (float*)d_ws;
  float* ecb     = ws;                 // 262144
  float* zq      = ws + 262144;        // 262144
  float* q       = ws + 524288;        // 262144
  float* qproj   = ws + 786432;        // 1048576
  float* ebar    = ws + 1835008;       // 1048576
  float* x1      = ws + 2883584;       // 262144
  float* x2      = ws + 3145728;       // 262144
  float* qbias   = ws + 3407872;       // 128
  float* bias_hz = ws + 3408000;       // 512
  unsigned short* ebf = (unsigned short*)(ws + 3408512); // e as bf16: 33.5M elems (67 MB)

  float* out_mean = (float*)d_out;            // 262144
  float* out_logv = out_mean + 262144;        // 262144
  float* out_a    = out_mean + 524288;        // 262144
  float* out_att  = out_mean + 786432;        // 16384

  dim3 blk(256);
  const long SN = (long)HIDC * PC;   // 8192

  vec_add2<<<2, blk, 0, stream>>>(bh, bz, bias_hz, HIDC);

  // ecb = Wc @ c + bc            (M=512, cols=32*16)
  wmma_gemm<0,0><<<dim3(4,4), blk, 0, stream>>>(Wc, cc, 512, nullptr, nullptr, 0,
      512, SN, bc, nullptr, 1, 0, ecb, SN);

  // e = bf16( Wh@h + Wz@z + (bh+bz) + ecb[b] )    (M=512, cols=4096*16)
  wmma_gemm<1,0><<<dim3(512,4), blk, 0, stream>>>(Wh, h, 512, Wz, z, 512,
      512, SN, bias_hz, ecb, NSC, 8192, ebf, SN);

  zq_reduce_kernel<<<1024, blk, 0, stream>>>(ebf, zq);

  // q = Wq @ zq + bq
  wmma_gemm<0,0><<<dim3(4,4), blk, 0, stream>>>(Wq, zq, 512, nullptr, nullptr, 0,
      512, SN, bq, nullptr, 1, 0, q, SN);

  qproj_kernel<<<4096, blk, 0, stream>>>(Wk, q, qproj);
  qbias_kernel<<<128, blk, 0, stream>>>(q, bk, qbias);
  sim_kernel<<<4096, blk, 0, stream>>>(ebf, qproj, qbias, out_att);
  softmax_kernel<<<128, dim3(128), 0, stream>>>(out_att);
  ebar_kernel<<<1024, blk, 0, stream>>>(ebf, out_att, ebar);

  // a_head = Wv_head @ ebar_head + bv_head   (M=128 per head, cols=32*16)
  for (int n = 0; n < NHEADC; ++n) {
    wmma_gemm<0,0><<<dim3(4,1), blk, 0, stream>>>(
        Wv + (long)n * HDC * HIDC, ebar + (long)n * 8192, 512,
        nullptr, nullptr, 0, 512, (long)32768,
        bv + n * HDC, nullptr, 1, 0, out_a + n * HDC * PC, SN);
  }

  // MLP: 3x elu(conv)
  wmma_gemm<0,1><<<dim3(4,4), blk, 0, stream>>>(Wp, out_a, 512, nullptr, nullptr, 0,
      512, SN, bp, nullptr, 1, 0, x1, SN);
  wmma_gemm<0,1><<<dim3(4,4), blk, 0, stream>>>(Wp + 262144, x1, 512, nullptr, nullptr, 0,
      512, SN, bp + 512, nullptr, 1, 0, x2, SN);
  wmma_gemm<0,1><<<dim3(4,4), blk, 0, stream>>>(Wp + 524288, x2, 512, nullptr, nullptr, 0,
      512, SN, bp + 1024, nullptr, 1, 0, x1, SN);

  // stats = Wo @ x + bo, split into mean / logvar
  wmma_gemm<0,0><<<dim3(4,4), blk, 0, stream>>>(Wo, x1, 512, nullptr, nullptr, 0,
      512, SN, bo, nullptr, 1, 0, out_mean, SN);
  wmma_gemm<0,0><<<dim3(4,4), blk, 0, stream>>>(Wo + 262144, x1, 512, nullptr, nullptr, 0,
      512, SN, bo + 512, nullptr, 1, 0, out_logv, SN);
}
